// MambaBlock_70111046140103
// MI455X (gfx1250) — matrix operand order repaired and checked by reading, not compiled, not run on hardware
//
#include <hip/hip_runtime.h>

// ---------------- problem constants (from reference) ----------------
#define BN_    2
#define LSEQ   1024
#define DM_    1024
#define DSTATE 16
#define DI     2048            // d_inner = EXPAND*DM
#define NROWS  (BN_*LSEQ)      // 2048 token rows

typedef __attribute__((ext_vector_type(16))) __bf16 v16bf;
typedef __attribute__((ext_vector_type(8)))  float  v8f;

// ---------------- workspace layout (bytes) ----------------
static constexpr size_t MB        = 1024ull * 1024ull;
static constexpr size_t OFF_XB    = 0;                    // x bf16           4 MB
static constexpr size_t OFF_W1T   = 4*MB;                 // in_w^T bf16      8 MB
static constexpr size_t OFF_DTWT  = 12*MB;                // dt_w^T bf16      8 MB
static constexpr size_t OFF_OUTWT = 20*MB;                // out_w^T bf16     4 MB
static constexpr size_t OFF_BCWT  = 24*MB;                // [B_w;C_w]^T bf16 128 KB
static constexpr size_t OFF_BCB   = 24*MB + 256*1024;     // fused bias       128 B
static constexpr size_t OFF_XP    = 25*MB;                // in-proj f32      32 MB
static constexpr size_t OFF_XCONV = 57*MB;                // silu(conv) f32   16 MB
static constexpr size_t OFF_XCB   = 73*MB;                // silu(conv) bf16  8 MB
static constexpr size_t OFF_DT    = 81*MB;                // softplus(dt) f32 16 MB
static constexpr size_t OFF_BC    = 97*MB;                // [Bm|Cm] f32      256 KB
static constexpr size_t OFF_Y     = 98*MB;                // scan out f32     16 MB
static constexpr size_t OFF_YB16  = 114*MB;               // gated y bf16     8 MB
// total 122 MB

// ---------------- helpers ----------------
__device__ __forceinline__ unsigned short f32_to_bf16(float f) {
  unsigned int u = __float_as_uint(f);
  u += 0x7FFFu + ((u >> 16) & 1u);           // round-to-nearest-even
  return (unsigned short)(u >> 16);
}
__device__ __forceinline__ float softplusf(float x) {
  return (x > 20.f) ? x : __logf(1.f + __expf(x));
}
__device__ __forceinline__ float siluf(float x) {
  return x / (1.f + __expf(-x));
}

// async global -> LDS copy, 16 bytes, tracked by ASYNCcnt (CDNA5)
__device__ __forceinline__ void async_copy_b128(const unsigned short* g, unsigned lds_byte_addr) {
  asm volatile("global_load_async_to_lds_b128 %0, %1, off"
               :: "v"(lds_byte_addr), "v"(g) : "memory");
}
__device__ __forceinline__ void wait_async0() {
  asm volatile("s_wait_asynccnt 0x0" ::: "memory");
}

// ---------------- conversion kernels ----------------
__global__ void cvt_bf16_kernel(const float* __restrict__ src,
                                unsigned short* __restrict__ dst, int n) {
  int i = blockIdx.x * blockDim.x + threadIdx.x;
  if (i < n) dst[i] = f32_to_bf16(src[i]);
}

// src is K x N (row-major), dst is N x K (row-major) = src^T, converted to bf16
__global__ void transpose_bf16_kernel(const float* __restrict__ src,
                                      unsigned short* __restrict__ dst,
                                      int K, int N) {
  int i = blockIdx.x * blockDim.x + threadIdx.x;
  if (i >= K * N) return;
  int k = i / N, n = i - k * N;
  dst[(size_t)n * K + k] = f32_to_bf16(src[i]);
}

// builds fused [B_w ; C_w]^T as bf16 [32][DI] plus fused bias[32]
__global__ void build_bc_kernel(const float* __restrict__ B_w,   // [DI][16]
                                const float* __restrict__ C_w,   // [DI][16]
                                const float* __restrict__ B_b,   // [16]
                                const float* __restrict__ C_b,   // [16]
                                unsigned short* __restrict__ bcwt,  // [32][DI]
                                float* __restrict__ bcbias) {       // [32]
  int i = blockIdx.x * blockDim.x + threadIdx.x;
  if (i >= 32 * DI) return;
  int row = i / DI, k = i - row * DI;
  float v = (row < 16) ? B_w[(size_t)k * 16 + row] : C_w[(size_t)k * 16 + (row - 16)];
  bcwt[(size_t)row * DI + k] = f32_to_bf16(v);
  if (i < 32) bcbias[i] = (i < 16) ? B_b[i] : C_b[i - 16];
}

union Frag { float4 f[2]; v16bf v; };

// =====================================================================
// LDS-staged double-buffered WMMA GEMM:  C[MxN] = A[MxK] * Bt[NxK]^T + bias
// Block = 256 threads (8 waves), tile 128M x 128N, wave subtile 32M x 64N.
// Global -> LDS via global_load_async_to_lds_b128 (ASYNCcnt), fragments via
// ds_load_b128. Requires M%128==0, N%128==0, K%32==0.
// act: 0 = none, 1 = softplus
// =====================================================================
__global__ __launch_bounds__(256) void gemm_bf16_wmma_lds(
    const unsigned short* __restrict__ A,   // M x K bf16 row-major
    const unsigned short* __restrict__ Bt,  // N x K bf16 row-major (B^T)
    const float* __restrict__ bias,         // N
    float* __restrict__ C,                  // M x N f32
    int M, int N, int K, int act) {
  // LDS: [buf][A|B][128 rows][40 ushorts] ; 40-ushort (80B) row pitch ->
  // 16-lane b128 fragment reads land on 16 distinct bank groups.
  __shared__ alignas(16) unsigned short sh[2][2][128 * 40];   // 40 KB

  const int tid  = threadIdx.x;
  const int lane = tid & 31;
  const int wave = tid >> 5;
  const int tilesM = M >> 7;
  const int bm = blockIdx.x % tilesM;
  const int bn = blockIdx.x / tilesM;
  const int m   = lane & 15;   // fragment row/col within 16x16 tile
  const int khi = lane >> 4;   // K-half selector
  const int wm  = wave & 3;    // wave M quadrant (32 rows)
  const int wn  = wave >> 2;   // wave N half (64 cols)

  const unsigned shbase = (unsigned)(uintptr_t)(&sh[0][0][0]);
  constexpr unsigned AB_STRIDE  = 128u * 40u * 2u;      // bytes per matrix slab
  constexpr unsigned BUF_STRIDE = 2u * AB_STRIDE;       // bytes per buffer

  // --- staging role of this thread: row r, 32-byte half `half` of a 64B row chunk
  const int r    = tid >> 1;
  const int half = tid & 1;
  const unsigned short* ga = A  + (size_t)(bm * 128 + r) * K + half * 16;
  const unsigned short* gb = Bt + (size_t)(bn * 128 + r) * K + half * 16;
  const unsigned ldsA = shbase + (unsigned)(r * 40 + half * 16) * 2u;
  const unsigned ldsB = ldsA + AB_STRIDE;

  auto stage = [&](int ks, int buf) {
    const unsigned short* pa = ga + ks * 32;
    const unsigned short* pb = gb + ks * 32;
    const unsigned boff = (unsigned)buf * BUF_STRIDE;
    async_copy_b128(pa,     ldsA + boff);
    async_copy_b128(pa + 8, ldsA + boff + 16u);
    async_copy_b128(pb,     ldsB + boff);
    async_copy_b128(pb + 8, ldsB + boff + 16u);
  };

  v8f acc[2][4] = {};

  const int ksteps = K >> 5;
  int buf = 0;
  stage(0, 0);

  for (int ks = 0; ks < ksteps; ++ks) {
    wait_async0();        // my async writes for `buf` complete
    __syncthreads();      // everyone's writes visible; prev compute done
    if (ks + 1 < ksteps)
      stage(ks + 1, buf ^ 1);   // DMA next tile under the WMMAs

    const unsigned short* shA = &sh[buf][0][0];
    const unsigned short* shB = &sh[buf][1][0];

    // A fragments (16x32 bf16): lane<16 -> K 0..7 & 16..23, lane>=16 -> 8..15 & 24..31
    Frag af[2];
#pragma unroll
    for (int rt = 0; rt < 2; ++rt) {
      const int row = wm * 32 + rt * 16 + m;
      af[rt].f[0] = *reinterpret_cast<const float4*>(shA + row * 40 + 8 * khi);
      af[rt].f[1] = *reinterpret_cast<const float4*>(shA + row * 40 + 16 + 8 * khi);
    }
#pragma unroll
    for (int ct = 0; ct < 4; ++ct) {
      // B fragment (32x16 bf16): lane<16 -> K 0..7 & 16..23, lane>=16 -> 8..15 & 24..31
      Frag bf;
      const int nrow = wn * 64 + ct * 16 + m;
      bf.f[0] = *reinterpret_cast<const float4*>(shB + nrow * 40 + 8 * khi);
      bf.f[1] = *reinterpret_cast<const float4*>(shB + nrow * 40 + 16 + 8 * khi);
#pragma unroll
      for (int rt = 0; rt < 2; ++rt)
        acc[rt][ct] = __builtin_amdgcn_wmma_f32_16x16x32_bf16(
            false, af[rt].v, false, bf.v, (short)0, acc[rt][ct], false, false);
    }
    buf ^= 1;
  }

  // Epilogue: C/D layout — lanes 0-15: M=rr, N=lane; lanes 16-31: M=8+rr, N=lane-16
#pragma unroll
  for (int rt = 0; rt < 2; ++rt)
#pragma unroll
    for (int ct = 0; ct < 4; ++ct) {
      const int col = bn * 128 + wn * 64 + ct * 16 + m;
      const float bv = bias ? bias[col] : 0.f;
#pragma unroll
      for (int rr = 0; rr < 8; ++rr) {
        const int row = bm * 128 + wm * 32 + rt * 16 + khi * 8 + rr;
        float v = acc[rt][ct][rr] + bv;
        if (act == 1) v = softplusf(v);
        C[(size_t)row * N + col] = v;
      }
    }
}

// ---------------- small-N direct WMMA GEMM (used for fused B/C, N=32) ----------------
template <int NT>
__global__ __launch_bounds__(256) void gemm_bf16_wmma(
    const unsigned short* __restrict__ A,   // M x K bf16 row-major
    const unsigned short* __restrict__ Bt,  // N x K bf16 row-major
    const float* __restrict__ bias,         // N
    float* __restrict__ C,                  // M x N f32
    int M, int N, int K, int act) {
  const int wave   = (blockIdx.x * blockDim.x + threadIdx.x) >> 5;
  const int lane   = threadIdx.x & 31;
  const int tilesM = M >> 4;
  const int tM     = wave % tilesM;
  const int tN     = wave / tilesM;
  const int m      = lane & 15;
  const int khi    = lane >> 4;

  const unsigned short* arow = A + (size_t)(tM * 16 + m) * K;
  const unsigned short* brow[NT];
#pragma unroll
  for (int i = 0; i < NT; ++i)
    brow[i] = Bt + (size_t)((tN * NT + i) * 16 + m) * K;

  v8f acc[NT] = {};
  const int ksteps = K >> 5;
  for (int ks = 0; ks < ksteps; ++ks) {
    Frag af;
    const int abase = ks * 32 + 8 * khi;
    af.f[0] = *reinterpret_cast<const float4*>(arow + abase);
    af.f[1] = *reinterpret_cast<const float4*>(arow + abase + 16);
    if (ks + 1 < ksteps)
      __builtin_prefetch(arow + abase + 32, 0, 3);
#pragma unroll
    for (int i = 0; i < NT; ++i) {
      Frag bf;
      const int bbase = ks * 32 + 8 * khi;
      bf.f[0] = *reinterpret_cast<const float4*>(brow[i] + bbase);
      bf.f[1] = *reinterpret_cast<const float4*>(brow[i] + bbase + 16);
      acc[i] = __builtin_amdgcn_wmma_f32_16x16x32_bf16(
          false, af.v, false, bf.v, (short)0, acc[i], false, false);
    }
  }
#pragma unroll
  for (int i = 0; i < NT; ++i) {
    const int ncol = (tN * NT + i) * 16 + m;
    const float bv = bias ? bias[ncol] : 0.f;
#pragma unroll
    for (int r = 0; r < 8; ++r) {
      const int mrow = tM * 16 + khi * 8 + r;
      float v = acc[i][r] + bv;
      if (act == 1) v = softplusf(v);
      C[(size_t)mrow * N + ncol] = v;
    }
  }
}

// ---------------- depthwise SAME conv (window=4, pad_low=1) + SiLU ----------------
__global__ void conv_silu_kernel(const float* __restrict__ xp,      // [NROWS][2*DI]
                                 const float* __restrict__ conv_w,  // [4][1][DI]
                                 const float* __restrict__ conv_b,  // [DI]
                                 float* __restrict__ xconv,         // [NROWS][DI]
                                 unsigned short* __restrict__ xcb)  // [NROWS][DI] bf16
{
  int idx = blockIdx.x * blockDim.x + threadIdx.x;
  if (idx >= NROWS * DI) return;
  int c   = idx % DI;
  int row = idx / DI;
  int t   = row % LSEQ;
  int b   = row / LSEQ;
  float s = conv_b[c];
#pragma unroll
  for (int k = 0; k < 4; ++k) {
    int tt = t - 1 + k;                                // SAME pad: low=1, high=2
    if (tt >= 0 && tt < LSEQ)
      s += conv_w[k * DI + c] * xp[((size_t)(b * LSEQ + tt)) * (2 * DI) + c];
  }
  float v = siluf(s);
  xconv[idx] = v;
  xcb[idx]   = f32_to_bf16(v);
}

// ---------------- selective scan: one thread per (b, d) row, 16 states in regs ----------------
__global__ __launch_bounds__(256) void scan_kernel(
    const float* __restrict__ dt,     // [NROWS][DI]   (softplus applied)
    const float* __restrict__ bc,     // [NROWS][32]   (Bm | Cm)
    const float* __restrict__ xconv,  // [NROWS][DI]
    const float* __restrict__ A,      // [DI][16]
    float* __restrict__ y)            // [NROWS][DI]
{
  int tid = blockIdx.x * blockDim.x + threadIdx.x;   // 0..BN_*DI-1
  if (tid >= BN_ * DI) return;
  int d = tid % DI;
  int b = tid / DI;
  float a[DSTATE], h[DSTATE];
#pragma unroll
  for (int s = 0; s < DSTATE; ++s) { a[s] = A[(size_t)d * DSTATE + s]; h[s] = 0.f; }

  for (int t = 0; t < LSEQ; ++t) {
    size_t row = (size_t)b * LSEQ + t;
    float dtv = dt[row * DI + d];
    float xv  = xconv[row * DI + d];
    const float* bcr = bc + row * 32;
    float acc = 0.f;
#pragma unroll
    for (int s = 0; s < DSTATE; ++s) {
      float hh = __expf(dtv * a[s]) * h[s] + dtv * bcr[s] * xv;   // dA*h + dB*x
      h[s] = hh;
      acc += hh * bcr[16 + s];                                    // sum h*C
    }
    y[row * DI + d] = acc;
  }
}

// ---------------- residual + gate, convert to bf16 ----------------
__global__ void gate_kernel(const float* __restrict__ y,       // [NROWS][DI]
                            const float* __restrict__ xconv,   // [NROWS][DI]
                            const float* __restrict__ Dp,      // [DI]
                            const float* __restrict__ xp,      // [NROWS][2*DI] (gate half)
                            unsigned short* __restrict__ yb16) // [NROWS][DI] bf16
{
  int idx = blockIdx.x * blockDim.x + threadIdx.x;
  if (idx >= NROWS * DI) return;
  int c   = idx % DI;
  int row = idx / DI;
  float g = xp[(size_t)row * (2 * DI) + DI + c];
  float v = (y[idx] + xconv[idx] * Dp[c]) * siluf(g);
  yb16[idx] = f32_to_bf16(v);
}

// ---------------- launch ----------------
extern "C" void kernel_launch(void* const* d_in, const int* in_sizes, int n_in,
                              void* d_out, int out_size, void* d_ws, size_t ws_size,
                              hipStream_t stream) {
  const float* x      = (const float*)d_in[0];   // [2,1024,1024]
  const float* in_w   = (const float*)d_in[1];   // [1024,4096]
  const float* in_b   = (const float*)d_in[2];   // [4096]
  const float* conv_w = (const float*)d_in[3];   // [4,1,2048]
  const float* conv_b = (const float*)d_in[4];   // [2048]
  const float* A      = (const float*)d_in[5];   // [2048,16]
  const float* Dp     = (const float*)d_in[6];   // [2048]
  const float* B_w    = (const float*)d_in[7];   // [2048,16]
  const float* B_b    = (const float*)d_in[8];   // [16]
  const float* C_w    = (const float*)d_in[9];   // [2048,16]
  const float* C_b    = (const float*)d_in[10];  // [16]
  const float* dt_w   = (const float*)d_in[11];  // [2048,2048]
  const float* dt_b   = (const float*)d_in[12];  // [2048]
  const float* out_w  = (const float*)d_in[13];  // [2048,1024]
  const float* out_b  = (const float*)d_in[14];  // [1024]

  char* ws = (char*)d_ws;
  unsigned short* xb    = (unsigned short*)(ws + OFF_XB);
  unsigned short* w1t   = (unsigned short*)(ws + OFF_W1T);
  unsigned short* dtwt  = (unsigned short*)(ws + OFF_DTWT);
  unsigned short* outwt = (unsigned short*)(ws + OFF_OUTWT);
  unsigned short* bcwt  = (unsigned short*)(ws + OFF_BCWT);
  float*          bcb   = (float*)(ws + OFF_BCB);
  float*          xp    = (float*)(ws + OFF_XP);
  float*          xconv = (float*)(ws + OFF_XCONV);
  unsigned short* xcb   = (unsigned short*)(ws + OFF_XCB);
  float*          dtb   = (float*)(ws + OFF_DT);
  float*          bcbuf = (float*)(ws + OFF_BC);
  float*          ybuf  = (float*)(ws + OFF_Y);
  unsigned short* yb16  = (unsigned short*)(ws + OFF_YB16);
  float*          out   = (float*)d_out;

  const int TPB = 256;

  // 1) precision/layout conversions
  cvt_bf16_kernel<<<(NROWS * DM_) / TPB, TPB, 0, stream>>>(x, xb, NROWS * DM_);
  transpose_bf16_kernel<<<(DM_ * 2 * DI) / TPB, TPB, 0, stream>>>(in_w, w1t, DM_, 2 * DI);
  transpose_bf16_kernel<<<(DI * DI) / TPB, TPB, 0, stream>>>(dt_w, dtwt, DI, DI);
  transpose_bf16_kernel<<<(DI * DM_) / TPB, TPB, 0, stream>>>(out_w, outwt, DI, DM_);
  build_bc_kernel<<<(32 * DI) / TPB, TPB, 0, stream>>>(B_w, C_w, B_b, C_b, bcwt, bcb);

  // 2) xp = x @ in_w + in_b   (M=2048, N=4096, K=1024) : 16x32 = 512 block tiles
  gemm_bf16_wmma_lds<<<(NROWS / 128) * ((2 * DI) / 128), TPB, 0, stream>>>(
      xb, w1t, in_b, xp, NROWS, 2 * DI, DM_, 0);

  // 3) depthwise conv + silu
  conv_silu_kernel<<<(NROWS * DI) / TPB, TPB, 0, stream>>>(xp, conv_w, conv_b, xconv, xcb);

  // 4) dt = softplus(x_conv @ dt_w + dt_b)   (M=2048, N=2048, K=2048)
  gemm_bf16_wmma_lds<<<(NROWS / 128) * (DI / 128), TPB, 0, stream>>>(
      xcb, dtwt, dt_b, dtb, NROWS, DI, DI, 1);

  // 5) [Bm|Cm] = x_conv @ [B_w|C_w] + bias   (M=2048, N=32, K=2048)
  gemm_bf16_wmma<2><<<128 / 8, TPB, 0, stream>>>(xcb, bcwt, bcb, bcbuf, NROWS, 32, DI, 0);

  // 6) selective scan
  scan_kernel<<<(BN_ * DI) / TPB, TPB, 0, stream>>>(dtb, bcbuf, xconv, A, ybuf);

  // 7) residual + gate -> bf16
  gate_kernel<<<(NROWS * DI) / TPB, TPB, 0, stream>>>(ybuf, xconv, Dp, xp, yb16);

  // 8) out = y @ out_w + out_b   (M=2048, N=1024, K=2048)
  gemm_bf16_wmma_lds<<<(NROWS / 128) * (DM_ / 128), TPB, 0, stream>>>(
      yb16, outwt, out_b, out, NROWS, DM_, DI, 0);
}
